// SubnetworkFeatureEmbedding_26834955665549
// MI455X (gfx1250) — matrix operand order repaired and checked by reading, not compiled, not run on hardware
//
#include <hip/hip_runtime.h>
#include <hip/hip_bf16.h>

typedef __attribute__((ext_vector_type(16))) __bf16 v16bf;
typedef __attribute__((ext_vector_type(8)))  float  v8f;

#define NSUB    8
#define REG     50
#define TOT     400
#define HID     512
#define FLATC   2500
#define KP1     2560          // 2500 padded to multiple of BK=64
#define BATCH   256
#define ROWS    (NSUB * BATCH)   // 2048

// ---------- helpers ----------
static __device__ __forceinline__ __bf16 f2bf(float f) {
    unsigned u = __builtin_bit_cast(unsigned, f);
    u += 0x7fffu + ((u >> 16) & 1u);          // round-to-nearest-even
    unsigned short h = (unsigned short)(u >> 16);
    return __builtin_bit_cast(__bf16, h);
}

// ---------- gather diagonal 50x50 blocks of x into padded bf16 A ----------
// flatA layout: [s][b][KP1] (subnet-major rows so batched GEMM is contiguous)
__global__ __launch_bounds__(256) void gather_bf16_kernel(
    const float* __restrict__ x, __bf16* __restrict__ flatA) {
    const int r = blockIdx.x;          // r = s*256 + b
    const int s = r >> 8;
    const int b = r & 255;
    const float* xb = x + (long)b * TOT * TOT;
    __bf16* out = flatA + (long)r * KP1;
    for (int f = threadIdx.x; f < KP1; f += 256) {
        float v = 0.0f;
        if (f < FLATC) {
            int i = f / REG, j = f - i * REG;
            v = xb[(long)(s * REG + i) * TOT + (s * REG + j)];
        }
        out[f] = f2bf(v);
    }
}

// ---------- tiled transpose + f32->bf16: dst[z][n][kp] = src[z][k][n] ----------
__global__ __launch_bounds__(256) void transpose_bf16_kernel(
    const float* __restrict__ src, __bf16* __restrict__ dst,
    int K, int N, int Kp) {
    __shared__ float tile[32][33];
    const int z = blockIdx.z;
    src += (long)z * K * N;
    dst += (long)z * N * Kp;
    const int k0 = blockIdx.x * 32;
    const int n0 = blockIdx.y * 32;
    const int tx = threadIdx.x;        // 0..31
    const int ty = threadIdx.y;        // 0..7
#pragma unroll
    for (int i = 0; i < 4; ++i) {
        int kk = k0 + ty + 8 * i;
        float v = (kk < K) ? src[(long)kk * N + (n0 + tx)] : 0.0f;
        tile[ty + 8 * i][tx] = v;
    }
    __syncthreads();
#pragma unroll
    for (int i = 0; i < 4; ++i) {
        int nn = n0 + ty + 8 * i;
        dst[(long)nn * Kp + (k0 + tx)] = f2bf(tile[tx][ty + 8 * i]);
    }
}

// ---------- batched bf16 WMMA GEMM ----------
// C[M,N] = act(A[M,K](bf16,row-major) * Bt[N,K](bf16,N-major) + bias) (+Res)
// block: 256 thr = 8 waves (2x4), wave tile 32x32 (2x2 wmma frags)
#define BM 64
#define BN 128
#define BK 64
#define LDT (BK + 8)    // 72 elems = 144B rows: 16B aligned, conflict-free frags

__global__ __launch_bounds__(256) void gemm_bf16_kernel(
    const __bf16* __restrict__ A, const __bf16* __restrict__ Bt,
    const float* __restrict__ bias, float* __restrict__ Cf,
    __bf16* __restrict__ Cb, const float* __restrict__ Res,
    int M, int N, int K,
    long sA, long sB, long sBias, long sC,
    int relu, int swapOut) {
    const int z = blockIdx.z;
    A  += (long)z * sA;
    Bt += (long)z * sB;
    const float* biasz = bias + (long)z * sBias;
    const long cOff = (long)z * sC;

    __shared__ __align__(16) __bf16 As[BM][LDT];
    __shared__ __align__(16) __bf16 Bs[BN][LDT];

    const int tid  = threadIdx.x;
    const int wave = tid >> 5;
    const int lane = tid & 31;
    const int wm   = wave >> 2;      // 0..1
    const int wn   = wave & 3;       // 0..3
    const int half = lane >> 4;      // 0..1
    const int l16  = lane & 15;

    const int bm0 = blockIdx.y * BM;
    const int bn0 = blockIdx.x * BN;

    v8f c[2][2];
#pragma unroll
    for (int mi = 0; mi < 2; ++mi)
#pragma unroll
        for (int ni = 0; ni < 2; ++ni)
#pragma unroll
            for (int e = 0; e < 8; ++e) c[mi][ni][e] = 0.0f;

    const int arow = tid >> 2;             // 0..63
    const int acol = (tid & 3) << 4;       // 0,16,32,48 (elems)
    const int brow = tid >> 1;             // 0..127
    const int bcol = (tid & 1) << 5;       // 0,32 (elems)

    for (int k0 = 0; k0 < K; k0 += BK) {
        const uint4* ag = (const uint4*)(A + (long)(bm0 + arow) * K + (k0 + acol));
        uint4 a0 = ag[0], a1 = ag[1];
        const uint4* bg = (const uint4*)(Bt + (long)(bn0 + brow) * K + (k0 + bcol));
        uint4 b0 = bg[0], b1 = bg[1], b2 = bg[2], b3 = bg[3];
        __syncthreads();
        *(uint4*)&As[arow][acol]      = a0;
        *(uint4*)&As[arow][acol + 8]  = a1;
        *(uint4*)&Bs[brow][bcol]      = b0;
        *(uint4*)&Bs[brow][bcol + 8]  = b1;
        *(uint4*)&Bs[brow][bcol + 16] = b2;
        *(uint4*)&Bs[brow][bcol + 24] = b3;
        __syncthreads();

#pragma unroll
        for (int ki = 0; ki < 2; ++ki) {
            v16bf af[2], bfr[2];
            // A frag (16x32 bf16): lane m=l16, elems i<8 -> k=8*half+i ; i>=8 -> k=16+8*half+(i-8)
#pragma unroll
            for (int mi = 0; mi < 2; ++mi) {
                const __bf16* p0 = &As[wm * 32 + mi * 16 + l16][ki * 32 + 8 * half];
                const __bf16* p1 = p0 + 16;
#pragma unroll
                for (int j = 0; j < 8; ++j) { af[mi][j] = p0[j]; af[mi][8 + j] = p1[j]; }
            }
            // B frag (32x16 bf16): lane n=l16, elems i<8 -> k=8*half+i ; i>=8 -> k=16+8*half+(i-8)
#pragma unroll
            for (int ni = 0; ni < 2; ++ni) {
                const __bf16* p = &Bs[wn * 32 + ni * 16 + l16][ki * 32 + 8 * half];
#pragma unroll
                for (int j = 0; j < 16; ++j) bfr[ni][j] = p[j < 8 ? j : j + 8];
            }
#pragma unroll
            for (int mi = 0; mi < 2; ++mi)
#pragma unroll
                for (int ni = 0; ni < 2; ++ni)
                    c[mi][ni] = __builtin_amdgcn_wmma_f32_16x16x32_bf16(
                        false, af[mi], false, bfr[ni], (short)0, c[mi][ni],
                        false, false);
        }
    }

    // epilogue: C frag elem e -> row = base + 8*half + e, col = base + l16
#pragma unroll
    for (int mi = 0; mi < 2; ++mi) {
#pragma unroll
        for (int ni = 0; ni < 2; ++ni) {
            const int col = bn0 + wn * 32 + ni * 16 + l16;
            const float bv = biasz[col];
#pragma unroll
            for (int e = 0; e < 8; ++e) {
                const int row = bm0 + wm * 32 + mi * 16 + half * 8 + e;
                float v = c[mi][ni][e] + bv;
                if (relu) v = fmaxf(v, 0.0f);
                if (Res) v += Res[cOff + (long)row * N + col];
                if (Cb) Cb[cOff + (long)row * N + col] = f2bf(v);
                if (Cf) {
                    long orow = swapOut ? (long)((row & 255) * NSUB + (row >> 8))
                                        : (long)row;
                    Cf[cOff + orow * N + col] = v;
                }
            }
        }
    }
}

// ---------- LayerNorm over HID=512, one block per row ----------
__global__ __launch_bounds__(256) void layernorm_kernel(
    const float* __restrict__ xs, const float* __restrict__ g,
    const float* __restrict__ bb, __bf16* __restrict__ xn) {
    const int r = blockIdx.x;
    const int t = threadIdx.x;
    const float* row = xs + (long)r * HID;
    float v0 = row[t], v1 = row[t + 256];
    __shared__ float rs[256], rq[256];
    rs[t] = v0 + v1;
    rq[t] = v0 * v0 + v1 * v1;
    __syncthreads();
    for (int o = 128; o > 0; o >>= 1) {
        if (t < o) { rs[t] += rs[t + o]; rq[t] += rq[t + o]; }
        __syncthreads();
    }
    const float mu   = rs[0] * (1.0f / HID);
    const float var  = rq[0] * (1.0f / HID) - mu * mu;
    const float rstd = rsqrtf(var + 1e-5f);
    xn[(long)r * HID + t]       = f2bf((v0 - mu) * rstd * g[t] + bb[t]);
    xn[(long)r * HID + t + 256] = f2bf((v1 - mu) * rstd * g[t + 256] + bb[t + 256]);
}

// ---------- tiny 8x8 masked-softmax attention: 1 block/sample, 1 wave/head ----------
__global__ __launch_bounds__(256) void attention_kernel(
    const float* __restrict__ qf, const float* __restrict__ kf,
    const float* __restrict__ vf, __bf16* __restrict__ ctx) {
    const int b = blockIdx.x;
    const int h = threadIdx.x >> 5;
    const int lane = threadIdx.x & 31;
    if (lane >= NSUB) return;
    const int m = lane;
    const long qoff = ((long)(m * BATCH + b)) * HID + h * 64;
    float qreg[64];
#pragma unroll
    for (int d = 0; d < 64; ++d) qreg[d] = qf[qoff + d];
    float sc[NSUB];
    float mx = -1e30f;
    for (int n = 0; n < NSUB; ++n) {
        const long koff = ((long)(n * BATCH + b)) * HID + h * 64;
        float acc = 0.0f;
#pragma unroll
        for (int d = 0; d < 64; ++d) acc += qreg[d] * kf[koff + d];
        acc *= 0.125f;                         // 1/sqrt(64)
        if (n == m) acc = 0.0f;                // scores * (1 - eye)
        sc[n] = acc;
        mx = fmaxf(mx, acc);
    }
    float sum = 0.0f;
    for (int n = 0; n < NSUB; ++n) { sc[n] = __expf(sc[n] - mx); sum += sc[n]; }
    const float inv = 1.0f / sum;
    for (int d = 0; d < 64; ++d) {
        float acc = 0.0f;
        for (int n = 0; n < NSUB; ++n)
            acc += sc[n] * vf[((long)(n * BATCH + b)) * HID + h * 64 + d];
        ctx[qoff + d] = f2bf(acc * inv);
    }
}

// ---------- launch ----------
extern "C" void kernel_launch(void* const* d_in, const int* in_sizes, int n_in,
                              void* d_out, int out_size, void* d_ws, size_t ws_size,
                              hipStream_t stream) {
    (void)in_sizes; (void)n_in; (void)out_size; (void)ws_size;
    const float* x    = (const float*)d_in[0];   // d_in[1] = masks (structure known)
    const float* W1   = (const float*)d_in[2];
    const float* b1   = (const float*)d_in[3];
    const float* W2   = (const float*)d_in[4];
    const float* b2   = (const float*)d_in[5];
    const float* ln_g = (const float*)d_in[6];
    const float* ln_b = (const float*)d_in[7];
    const float* Wq   = (const float*)d_in[8];
    const float* bq   = (const float*)d_in[9];
    const float* Wk   = (const float*)d_in[10];
    const float* bk   = (const float*)d_in[11];
    const float* Wv   = (const float*)d_in[12];
    const float* bv   = (const float*)d_in[13];
    const float* Wo   = (const float*)d_in[14];
    const float* bo   = (const float*)d_in[15];
    const float* Wp   = (const float*)d_in[16];
    const float* bp   = (const float*)d_in[17];

    char* ws = (char*)d_ws;
    size_t off = 0;
    auto alloc = [&](size_t bytes) -> void* {
        void* p = ws + off;
        off += (bytes + 255) & ~(size_t)255;
        return p;
    };
    __bf16* flatA = (__bf16*)alloc((size_t)ROWS * KP1 * 2);
    __bf16* W1t   = (__bf16*)alloc((size_t)NSUB * HID * KP1 * 2);
    __bf16* hbuf  = (__bf16*)alloc((size_t)ROWS * HID * 2);
    __bf16* W2t   = (__bf16*)alloc((size_t)NSUB * HID * HID * 2);
    float*  xsub  = (float*) alloc((size_t)ROWS * HID * 4);
    __bf16* xn    = (__bf16*)alloc((size_t)ROWS * HID * 2);
    __bf16* Wqt   = (__bf16*)alloc((size_t)HID * HID * 2);
    __bf16* Wkt   = (__bf16*)alloc((size_t)HID * HID * 2);
    __bf16* Wvt   = (__bf16*)alloc((size_t)HID * HID * 2);
    __bf16* Wot   = (__bf16*)alloc((size_t)HID * HID * 2);
    __bf16* Wpt   = (__bf16*)alloc((size_t)HID * HID * 2);
    float*  qf    = (float*) alloc((size_t)ROWS * HID * 4);
    float*  kfb   = (float*) alloc((size_t)ROWS * HID * 4);
    float*  vfb   = (float*) alloc((size_t)ROWS * HID * 4);
    __bf16* ctx   = (__bf16*)alloc((size_t)ROWS * HID * 2);
    __bf16* t1    = (__bf16*)alloc((size_t)ROWS * HID * 2);

    const dim3 tb(32, 8);
    gather_bf16_kernel<<<ROWS, 256, 0, stream>>>(x, flatA);
    transpose_bf16_kernel<<<dim3(KP1/32, HID/32, NSUB), tb, 0, stream>>>(W1, W1t, FLATC, HID, KP1);
    transpose_bf16_kernel<<<dim3(HID/32, HID/32, NSUB), tb, 0, stream>>>(W2, W2t, HID, HID, HID);
    transpose_bf16_kernel<<<dim3(HID/32, HID/32, 1), tb, 0, stream>>>(Wq, Wqt, HID, HID, HID);
    transpose_bf16_kernel<<<dim3(HID/32, HID/32, 1), tb, 0, stream>>>(Wk, Wkt, HID, HID, HID);
    transpose_bf16_kernel<<<dim3(HID/32, HID/32, 1), tb, 0, stream>>>(Wv, Wvt, HID, HID, HID);
    transpose_bf16_kernel<<<dim3(HID/32, HID/32, 1), tb, 0, stream>>>(Wo, Wot, HID, HID, HID);
    transpose_bf16_kernel<<<dim3(HID/32, HID/32, 1), tb, 0, stream>>>(Wp, Wpt, HID, HID, HID);

    // layer 1: h = relu(flat @ W1 + b1)  [per-subnet, M=256,N=512,K=2560]
    gemm_bf16_kernel<<<dim3(HID/BN, BATCH/BM, NSUB), 256, 0, stream>>>(
        flatA, W1t, b1, nullptr, hbuf, nullptr,
        BATCH, HID, KP1, (long)BATCH*KP1, (long)HID*KP1, HID, (long)BATCH*HID, 1, 0);
    // layer 2: x_sub = h @ W2 + b2
    gemm_bf16_kernel<<<dim3(HID/BN, BATCH/BM, NSUB), 256, 0, stream>>>(
        hbuf, W2t, b2, xsub, nullptr, nullptr,
        BATCH, HID, HID, (long)BATCH*HID, (long)HID*HID, HID, (long)BATCH*HID, 0, 0);
    layernorm_kernel<<<ROWS, 256, 0, stream>>>(xsub, ln_g, ln_b, xn);
    // q/k/v projections, M=2048
    gemm_bf16_kernel<<<dim3(HID/BN, ROWS/BM, 1), 256, 0, stream>>>(
        xn, Wqt, bq, qf, nullptr, nullptr, ROWS, HID, HID, 0, 0, 0, 0, 0, 0);
    gemm_bf16_kernel<<<dim3(HID/BN, ROWS/BM, 1), 256, 0, stream>>>(
        xn, Wkt, bk, kfb, nullptr, nullptr, ROWS, HID, HID, 0, 0, 0, 0, 0, 0);
    gemm_bf16_kernel<<<dim3(HID/BN, ROWS/BM, 1), 256, 0, stream>>>(
        xn, Wvt, bv, vfb, nullptr, nullptr, ROWS, HID, HID, 0, 0, 0, 0, 0, 0);
    attention_kernel<<<BATCH, 256, 0, stream>>>(qf, kfb, vfb, ctx);
    // t1 = ctx @ Wo + bo
    gemm_bf16_kernel<<<dim3(HID/BN, ROWS/BM, 1), 256, 0, stream>>>(
        ctx, Wot, bo, nullptr, t1, nullptr, ROWS, HID, HID, 0, 0, 0, 0, 0, 0);
    // out = t1 @ Wp + bp + x_sub, rows permuted (s-major -> b-major) into d_out
    gemm_bf16_kernel<<<dim3(HID/BN, ROWS/BM, 1), 256, 0, stream>>>(
        t1, Wpt, bp, (float*)d_out, nullptr, xsub, ROWS, HID, HID, 0, 0, 0, 0, 0, 1);
}
